// MultiHeadAttention_85186381349424
// MI455X (gfx1250) — compile-verified, matrix operand order checked by reading, not run on hardware
//
#include <hip/hip_runtime.h>
#include <hip/hip_bf16.h>

// ---------------- problem constants ----------------
#define BB   4
#define SSQ  2048           // sequence length
#define HHID 1024           // hidden
#define NHH  16             // heads
#define HDD  64             // head dim

typedef __bf16 bf16;
typedef __attribute__((ext_vector_type(16))) __bf16 v16bf;
typedef __attribute__((ext_vector_type(8)))  __bf16 v8bf;
typedef __attribute__((ext_vector_type(4)))  __bf16 v4bf;
typedef __attribute__((ext_vector_type(8)))  float  v8f;
typedef __attribute__((ext_vector_type(4)))  float  v4f;

static constexpr size_t NTOK   = (size_t)BB * SSQ;      // 8192 tokens
static constexpr size_t SZ_ACT = NTOK * HHID;           // 8,388,608
static constexpr size_t SZ_W   = (size_t)HHID * HHID;   // 1,048,576

// ---------------- WMMA helpers ----------------
// ISA 16-bit A-matrix 16x32 packing (CDNA5 ISA §7.12.2):
//   lane L<16 : row M=L, K = {0..7, 16..23}
//   lane L>=16: row M=L-16, K = {8..15, 24..31}
// B operand is supplied as B^T rows in the same packing.
__device__ __forceinline__ v16bf load_bf_tile(const bf16* p, int ld, int lane) {
    const bf16* r = p + (size_t)(lane & 15) * ld + ((lane >> 4) << 3);
    v8bf lo = *(const v8bf*)(r);        // K = base+0..7   (16B)
    v8bf hi = *(const v8bf*)(r + 16);   // K = base+16..23 (16B)
    return __builtin_shufflevector(lo, hi, 0,1,2,3,4,5,6,7,8,9,10,11,12,13,14,15);
}

__device__ __forceinline__ v16bf load_f32_tile_bf(const float* p, int ld, int lane) {
    const float* r = p + (size_t)(lane & 15) * ld + ((lane >> 4) << 3);
    v4f a0 = *(const v4f*)(r);
    v4f a1 = *(const v4f*)(r + 4);
    v4f b0 = *(const v4f*)(r + 16);
    v4f b1 = *(const v4f*)(r + 20);
    v16bf v;
#pragma unroll
    for (int i = 0; i < 4; ++i) {
        v[i]      = (bf16)a0[i];
        v[4 + i]  = (bf16)a1[i];
        v[8 + i]  = (bf16)b0[i];
        v[12 + i] = (bf16)b1[i];
    }
    return v;
}

__device__ __forceinline__ v8f wmma_bf16(v16bf a, v16bf b, v8f c) {
    // (neg_a, A, neg_b, B, c_mod, C, reuse_a, reuse_b)
    return __builtin_amdgcn_wmma_f32_16x16x32_bf16(false, a, false, b, (short)0, c, false, false);
}

// ---------------- fp32 -> bf16 convert ----------------
__global__ void cvt_f32_bf16_kernel(const float* __restrict__ in, bf16* __restrict__ out, size_t n) {
    size_t i = ((size_t)blockIdx.x * blockDim.x + threadIdx.x) * 4;
    if (i >= n) return;
    v4f v = *(const v4f*)(in + i);
    v4bf o;
#pragma unroll
    for (int j = 0; j < 4; ++j) o[j] = (bf16)v[j];
    *(v4bf*)(out + i) = o;
}

// ---------------- generic 32x64-per-wave bf16 WMMA GEMM ----------------
// Two 16-row A sub-tiles share four B tiles: 8 WMMAs per 12 b128 loads per K-step.
// MODE 0: store bf16 row-major (ldc = HHID)          -> Q, K
// MODE 1: store bf16 transposed per head [b,h,d,s]   -> V^T
// MODE 2: store fp32 row-major (ldc = HHID)          -> O-proj
template <int MODE>
__global__ void gemm32x64_kernel(const bf16* __restrict__ A, int lda,
                                 const bf16* __restrict__ Bt, int ldb,
                                 int ksteps, int row_tiles, int col_tiles,
                                 void* __restrict__ Out) {
    int lane = threadIdx.x & 31;
    size_t wave = (size_t)blockIdx.x * (blockDim.x >> 5) + (threadIdx.x >> 5);
    if (wave >= (size_t)row_tiles * col_tiles) return;
    int ct = (int)(wave % col_tiles);
    int rt = (int)(wave / col_tiles);
    int rowbase = rt * 32, colbase = ct * 64;

    const bf16* Ab = A  + (size_t)rowbase * lda;
    const bf16* Bb = Bt + (size_t)colbase * ldb;
    v8f acc[2][4] = {};
    for (int kk = 0; kk < ksteps; ++kk) {
        if (kk + 1 < ksteps) {                       // pull next tiles toward WGP
            __builtin_prefetch(Ab + (size_t)(kk + 1) * 32, 0, 0);
            __builtin_prefetch(Bb + (size_t)(kk + 1) * 32, 0, 0);
        }
        v16bf a0 = load_bf_tile(Ab + kk * 32, lda, lane);
        v16bf a1 = load_bf_tile(Ab + (size_t)16 * lda + kk * 32, lda, lane);
#pragma unroll
        for (int n = 0; n < 4; ++n) {
            v16bf b = load_bf_tile(Bb + (size_t)(n * 16) * ldb + kk * 32, ldb, lane);
            acc[0][n] = wmma_bf16(a0, b, acc[0][n]);
            acc[1][n] = wmma_bf16(a1, b, acc[1][n]);
        }
    }
    // C/D layout: VGPR r, lane L -> (M = r + 8*(L>=16), N = L&15)
    int r0 = (lane >> 4) * 8, c = lane & 15;
    if (MODE == 0) {
        bf16* o = (bf16*)Out;
#pragma unroll
        for (int ms = 0; ms < 2; ++ms)
#pragma unroll
            for (int r = 0; r < 8; ++r) {
                size_t m = (size_t)rowbase + ms * 16 + r + r0;
#pragma unroll
                for (int n = 0; n < 4; ++n)
                    o[m * HHID + colbase + n * 16 + c] = (bf16)acc[ms][n][r];
            }
    } else if (MODE == 1) {
        bf16* o = (bf16*)Out;
        int h = colbase >> 6;
#pragma unroll
        for (int ms = 0; ms < 2; ++ms)
#pragma unroll
            for (int r = 0; r < 8; ++r) {
                size_t tok = (size_t)rowbase + ms * 16 + r + r0;
                size_t b_ = tok >> 11;            // / SSQ
                size_t s_ = tok & (SSQ - 1);
#pragma unroll
                for (int n = 0; n < 4; ++n) {
                    int d = n * 16 + c;
                    o[(((b_ * NHH + h) * HDD) + d) * (size_t)SSQ + s_] = (bf16)acc[ms][n][r];
                }
            }
    } else {
        float* o = (float*)Out;
#pragma unroll
        for (int ms = 0; ms < 2; ++ms)
#pragma unroll
            for (int r = 0; r < 8; ++r) {
                size_t m = (size_t)rowbase + ms * 16 + r + r0;
#pragma unroll
                for (int n = 0; n < 4; ++n)
                    o[m * HHID + colbase + n * 16 + c] = acc[ms][n][r];
            }
    }
}

// ---------------- scores = (Q K^T) / sqrt(H), causal mask, 32x64 per wave ----------------
__global__ void scores_kernel(const bf16* __restrict__ Qb, const bf16* __restrict__ Kb,
                              float* __restrict__ attn) {
    int lane = threadIdx.x & 31;
    size_t wave = (size_t)blockIdx.x * (blockDim.x >> 5) + (threadIdx.x >> 5);
    // waves: bh(64) x qt(64) x kt(32)
    int kt = (int)(wave & 31);
    int qt = (int)((wave >> 5) & 63);
    int bh = (int)(wave >> 11);
    int b = bh >> 4, h = bh & 15;
    int qbase = qt * 32, kbase = kt * 64;
    float* orow = attn + ((size_t)bh * SSQ + qbase) * SSQ + kbase;
    int r0 = (lane >> 4) * 8, c = lane & 15;

    if (kbase > qbase + 31) {   // tile fully above diagonal: uniform mask fill
#pragma unroll
        for (int ms = 0; ms < 2; ++ms)
#pragma unroll
            for (int r = 0; r < 8; ++r)
#pragma unroll
                for (int n = 0; n < 4; ++n)
                    orow[(size_t)(ms * 16 + r + r0) * SSQ + n * 16 + c] = -1e9f;
        return;
    }
    const bf16* Qt = Qb + ((size_t)b * SSQ + qbase) * HHID + h * HDD;
    const bf16* Kt = Kb + ((size_t)b * SSQ + kbase) * HHID + h * HDD;
    v8f acc[2][4] = {};
#pragma unroll
    for (int kk = 0; kk < 2; ++kk) {            // HD = 64 -> 2 WMMA K-steps
        v16bf a0 = load_bf_tile(Qt + kk * 32, HHID, lane);
        v16bf a1 = load_bf_tile(Qt + (size_t)16 * HHID + kk * 32, HHID, lane);
#pragma unroll
        for (int n = 0; n < 4; ++n) {
            v16bf bm = load_bf_tile(Kt + (size_t)(n * 16) * HHID + kk * 32, HHID, lane);
            acc[0][n] = wmma_bf16(a0, bm, acc[0][n]);
            acc[1][n] = wmma_bf16(a1, bm, acc[1][n]);
        }
    }
    const float scale = 0.03125f;               // 1/sqrt(1024)
#pragma unroll
    for (int ms = 0; ms < 2; ++ms)
#pragma unroll
        for (int r = 0; r < 8; ++r) {
            int q = qbase + ms * 16 + r + r0;
#pragma unroll
            for (int n = 0; n < 4; ++n) {
                int k = kbase + n * 16 + c;
                float v = acc[ms][n][r] * scale;
                orow[(size_t)(ms * 16 + r + r0) * SSQ + n * 16 + c] = (k > q) ? -1e9f : v;
            }
        }
}

// ---------------- row softmax in place ----------------
__global__ void softmax_kernel(float* __restrict__ attn) {
    __shared__ float red[256];
    float* p = attn + (size_t)blockIdx.x * SSQ;
    int t = threadIdx.x;
    float vals[8];
    float m = -3.4e38f;
#pragma unroll
    for (int i = 0; i < 8; ++i) { vals[i] = p[t + i * 256]; m = fmaxf(m, vals[i]); }
    red[t] = m; __syncthreads();
    for (int s = 128; s > 0; s >>= 1) { if (t < s) red[t] = fmaxf(red[t], red[t + s]); __syncthreads(); }
    m = red[0]; __syncthreads();
    float sum = 0.f;
#pragma unroll
    for (int i = 0; i < 8; ++i) { vals[i] = __expf(vals[i] - m); sum += vals[i]; }
    red[t] = sum; __syncthreads();
    for (int s = 128; s > 0; s >>= 1) { if (t < s) red[t] += red[t + s]; __syncthreads(); }
    float inv = 1.0f / red[0];
#pragma unroll
    for (int i = 0; i < 8; ++i) p[t + i * 256] = vals[i] * inv;
}

// ---------------- context = attn @ V (causally truncated K loop), 32x64 per wave ----------------
__global__ void av_kernel(const float* __restrict__ attn, const bf16* __restrict__ Vt,
                          bf16* __restrict__ ctx) {
    int lane = threadIdx.x & 31;
    size_t wave = (size_t)blockIdx.x * (blockDim.x >> 5) + (threadIdx.x >> 5);
    int qt = (int)(wave & 63);
    int bh = (int)(wave >> 6);
    int b = bh >> 4, h = bh & 15;
    int qbase = qt * 32;
    const float* Ab = attn + ((size_t)bh * SSQ + qbase) * SSQ;
    const bf16*  Bb = Vt + (size_t)bh * HDD * SSQ;
    int ksteps = qt + 1;                        // covers k <= qbase+31; attn is 0 beyond: exact
    v8f acc[2][4] = {};
    for (int kk = 0; kk < ksteps; ++kk) {
        if (kk + 1 < ksteps) {
            __builtin_prefetch(Ab + (size_t)(kk + 1) * 32, 0, 0);
            __builtin_prefetch(Bb + (size_t)(kk + 1) * 32, 0, 0);
        }
        v16bf a0 = load_f32_tile_bf(Ab + kk * 32, SSQ, lane);
        v16bf a1 = load_f32_tile_bf(Ab + (size_t)16 * SSQ + kk * 32, SSQ, lane);
#pragma unroll
        for (int n = 0; n < 4; ++n) {
            v16bf bm = load_bf_tile(Bb + (size_t)(n * 16) * SSQ + kk * 32, SSQ, lane);
            acc[0][n] = wmma_bf16(a0, bm, acc[0][n]);
            acc[1][n] = wmma_bf16(a1, bm, acc[1][n]);
        }
    }
    int r0 = (lane >> 4) * 8, c = lane & 15;
#pragma unroll
    for (int ms = 0; ms < 2; ++ms)
#pragma unroll
        for (int r = 0; r < 8; ++r) {
            size_t tok = (size_t)b * SSQ + qbase + ms * 16 + r + r0;
#pragma unroll
            for (int n = 0; n < 4; ++n)
                ctx[tok * HHID + h * HDD + n * 16 + c] = (bf16)acc[ms][n][r];
        }
}

// ---------------- residual + LayerNorm ----------------
__global__ void ln_kernel(const float* __restrict__ xo, const float* __restrict__ enc,
                          const float* __restrict__ gamma, const float* __restrict__ beta,
                          float* __restrict__ y) {
    __shared__ float red[256];
    size_t row = blockIdx.x;
    int t = threadIdx.x;
    const float* xr = xo + row * HHID;
    const float* er = enc + row * HHID;
    float x[4];
    float s = 0.f;
#pragma unroll
    for (int i = 0; i < 4; ++i) { x[i] = xr[t + i * 256] + er[t + i * 256]; s += x[i]; }
    red[t] = s; __syncthreads();
    for (int st = 128; st > 0; st >>= 1) { if (t < st) red[t] += red[t + st]; __syncthreads(); }
    float mu = red[0] * (1.0f / HHID); __syncthreads();
    float v = 0.f;
#pragma unroll
    for (int i = 0; i < 4; ++i) { float d = x[i] - mu; v += d * d; }
    red[t] = v; __syncthreads();
    for (int st = 128; st > 0; st >>= 1) { if (t < st) red[t] += red[t + st]; __syncthreads(); }
    float rstd = rsqrtf(red[0] * (1.0f / HHID) + 1e-6f);
    float* yr = y + row * HHID;
#pragma unroll
    for (int i = 0; i < 4; ++i) {
        int col = t + i * 256;
        yr[col] = (x[i] - mu) * rstd * gamma[col] + beta[col];
    }
}

// ---------------- host launch ----------------
extern "C" void kernel_launch(void* const* d_in, const int* in_sizes, int n_in,
                              void* d_out, int out_size, void* d_ws, size_t ws_size,
                              hipStream_t stream) {
    const float* enc   = (const float*)d_in[0];
    // d_in[1] = mask (causal tril; encoded analytically in scores_kernel)
    const float* WQ    = (const float*)d_in[2];
    const float* WK    = (const float*)d_in[3];
    const float* WV    = (const float*)d_in[4];
    const float* WO    = (const float*)d_in[5];
    const float* gamma = (const float*)d_in[6];
    const float* beta  = (const float*)d_in[7];

    float* y    = (float*)d_out;                 // [B,S,H]
    float* attn = y + SZ_ACT;                    // [B,NH,S,S]

    // workspace layout (~126 MB)
    bf16* enc_bf = (bf16*)d_ws;
    bf16* q_bf   = enc_bf + SZ_ACT;
    bf16* k_bf   = q_bf   + SZ_ACT;
    bf16* vt_bf  = k_bf   + SZ_ACT;              // [B,NH,HD,S]
    bf16* ctx_bf = vt_bf  + SZ_ACT;
    bf16* wq_bf  = ctx_bf + SZ_ACT;
    bf16* wk_bf  = wq_bf  + SZ_W;
    bf16* wv_bf  = wk_bf  + SZ_W;
    bf16* wo_bf  = wv_bf  + SZ_W;
    float* xo    = (float*)(wo_bf + SZ_W);       // fp32 [B*S, H]

    // 1) converts
    cvt_f32_bf16_kernel<<<(unsigned)(SZ_ACT / 4 / 256), 256, 0, stream>>>(enc, enc_bf, SZ_ACT);
    cvt_f32_bf16_kernel<<<(unsigned)(SZ_W / 4 / 256), 256, 0, stream>>>(WQ, wq_bf, SZ_W);
    cvt_f32_bf16_kernel<<<(unsigned)(SZ_W / 4 / 256), 256, 0, stream>>>(WK, wk_bf, SZ_W);
    cvt_f32_bf16_kernel<<<(unsigned)(SZ_W / 4 / 256), 256, 0, stream>>>(WV, wv_bf, SZ_W);
    cvt_f32_bf16_kernel<<<(unsigned)(SZ_W / 4 / 256), 256, 0, stream>>>(WO, wo_bf, SZ_W);

    // 2) projections: row_tiles = 8192/32 = 256, col_tiles = 1024/64 = 16 -> 4096 waves
    const unsigned projBlocks = 4096 / 8;        // 8 waves per 256-thread block
    gemm32x64_kernel<0><<<projBlocks, 256, 0, stream>>>(enc_bf, HHID, wq_bf, HHID, 32, 256, 16, q_bf);
    gemm32x64_kernel<0><<<projBlocks, 256, 0, stream>>>(enc_bf, HHID, wk_bf, HHID, 32, 256, 16, k_bf);
    gemm32x64_kernel<1><<<projBlocks, 256, 0, stream>>>(enc_bf, HHID, wv_bf, HHID, 32, 256, 16, vt_bf);

    // 3) scores: 64 bh * 64 qtiles * 32 ktiles = 131072 waves
    scores_kernel<<<131072 / 8, 256, 0, stream>>>(q_bf, k_bf, attn);

    // 4) softmax: one block per attention row (B*NH*S = 131072)
    softmax_kernel<<<131072u, 256, 0, stream>>>(attn);

    // 5) attn @ V: 64 bh * 64 qtiles = 4096 waves
    av_kernel<<<4096 / 8, 256, 0, stream>>>(attn, vt_bf, ctx_bf);

    // 6) output projection -> fp32
    gemm32x64_kernel<2><<<projBlocks, 256, 0, stream>>>(ctx_bf, HHID, wo_bf, HHID, 32, 256, 16, xo);

    // 7) residual + LN
    ln_kernel<<<(unsigned)NTOK, 256, 0, stream>>>(xo, enc, gamma, beta, y);
}
